// NodeConvolve_25786983645317
// MI455X (gfx1250) — compile-verified, matrix operand order checked by reading, not run on hardware
//
#include <hip/hip_runtime.h>

typedef __attribute__((ext_vector_type(16))) __bf16 v16bf;
typedef __attribute__((ext_vector_type(8)))  float  v8f;

#define NMOL 512
#define MA   64
#define FEAT 128
#define NF   16
#define DIN  2176   // FEAT*(1+NF)
#define H1   512
#define UPD  0.1f

// ---- helpers ---------------------------------------------------------------

// fp32 -> bf16 via hardware convert (v_cvt_pk_bf16_f32 class, single VALU op).
__device__ __forceinline__ unsigned short f2bf(float f) {
  return __builtin_bit_cast(unsigned short, (__bf16)f);
}

__device__ __forceinline__ unsigned pack2bf(float lo, float hi) {
  return (unsigned)f2bf(lo) | ((unsigned)f2bf(hi) << 16);
}

struct U4x2 { uint4 a, b; };

// Build a 16x32 bf16 WMMA operand fragment from LDS.
// Per ISA 7.12.2, lane<16 holds K {0..7,16..23}, lane>=16 holds K {8..15,24..31}
// -> two contiguous 16B chunks per lane at rowbase+khalf, rowbase+khalf+16.
__device__ __forceinline__ v16bf load_frag(const unsigned short* rowbase, int lane) {
  const int khalf = (lane >> 4) << 3;   // 0 or 8
  U4x2 p;
  p.a = *(const uint4*)(rowbase + khalf);
  p.b = *(const uint4*)(rowbase + khalf + 16);
  return __builtin_bit_cast(v16bf, p);
}

// LDS byte offset of a generic pointer (ISA: LDS aperture addr[31:0] = offset).
__device__ __forceinline__ unsigned lds_off(const void* p) {
  return (unsigned)(unsigned long long)p;
}

// Async global->LDS 16B copy per lane (GVS mode: SGPR64 base + VGPR u32 offset).
// Tracked by ASYNCcnt; no VGPR staging.
__device__ __forceinline__ void async_b128(unsigned lds_byte, unsigned gbyte,
                                           const void* base) {
  asm volatile("global_load_async_to_lds_b128 %0, %1, %2"
               :: "v"(lds_byte), "v"(gbyte), "s"(base) : "memory");
}

__device__ __forceinline__ void wait_async0() {
  asm volatile("s_wait_asynccnt 0x0" ::: "memory");
}

// ---- kernel 0: weight conversion fp32 -> bf16, TRANSPOSED ------------------
// w1t[n][k] = W1[k][n]  (H1 x DIN),  w2t[c][k] = W2[k][c]  (FEAT x H1)

__global__ void convert_weights(const float* __restrict__ W1,
                                const float* __restrict__ W2,
                                unsigned short* __restrict__ w1t,
                                unsigned short* __restrict__ w2t) {
  long i = (long)blockIdx.x * 256 + threadIdx.x;
  const long n1 = (long)DIN * H1;
  const long n2 = (long)H1 * FEAT;
  if (i < n1) {
    long n = i / DIN, k = i % DIN;
    w1t[i] = f2bf(W1[k * H1 + n]);
  } else {
    long j = i - n1;
    if (j < n2) {
      long c = j / H1, k = j % H1;
      w2t[j] = f2bf(W2[k * FEAT + c]);
    }
  }
}

// ---- kernel 1: RBF weights + neighbor aggregation (per molecule) -----------
// For each of 16 filters: u_f = w_f(64x64) @ h(64x128) via bf16 WMMA.
// Writes v = [h | u] (bf16) row-major (DIN=2176 cols).

__global__ void rbf_agg(const int*   __restrict__ z,
                        const float* __restrict__ r,
                        const float* __restrict__ h,
                        const float* __restrict__ dist,
                        const float* __restrict__ width,
                        unsigned short* __restrict__ vbuf) {
  __shared__ __align__(16) unsigned short sh_ht[FEAT * MA]; // B transposed: [feat][atom]
  __shared__ __align__(16) unsigned short sh_w[MA * MA];    // A: [atom a][atom b]
  __shared__ float sh_d[MA * MA];
  __shared__ float sh_r[MA * 3];
  __shared__ float sh_mask[MA];
  __shared__ float sh_mu[NF], sh_isig[NF];

  const int n    = blockIdx.x;
  const int tid  = threadIdx.x;
  const int wave = tid >> 5, lane = tid & 31;

  if (tid < MA)     sh_mask[tid] = (z[n * MA + tid] > -1) ? 1.f : 0.f;
  if (tid < MA * 3) sh_r[tid]    = r[n * MA * 3 + tid];
  if (tid < NF) { sh_mu[tid] = dist[tid]; sh_isig[tid] = 1.f / width[tid]; }

  // h: read pairs of atoms per feature -> packed b32 transposed LDS stores,
  // plus the h-part of v (coalesced across lanes: consecutive c).
  const float* hbase = h + (size_t)n * MA * FEAT;
  #pragma unroll
  for (int q = 0; q < (MA * FEAT) / 512; ++q) {
    int li = q * 256 + tid;            // 0..4095
    int c  = li & (FEAT - 1);
    int b0 = (li >> 7) << 1;           // even atom index
    unsigned short s0 = f2bf(hbase[(size_t)b0 * FEAT + c]);
    unsigned short s1 = f2bf(hbase[(size_t)(b0 + 1) * FEAT + c]);
    *(unsigned*)(sh_ht + c * MA + b0) = (unsigned)s0 | ((unsigned)s1 << 16);
    vbuf[((size_t)n * MA + b0) * DIN + c]     = s0;
    vbuf[((size_t)n * MA + b0 + 1) * DIN + c] = s1;
  }
  __syncthreads();

  // pairwise distance; fold pair mask in: masked pair -> huge d -> exp == 0
  #pragma unroll
  for (int q = 0; q < (MA * MA) / 256; ++q) {
    int idx = q * 256 + tid;
    int a = idx >> 6, b = idx & 63;
    float dx = sh_r[a * 3 + 0] - sh_r[b * 3 + 0];
    float dy = sh_r[a * 3 + 1] - sh_r[b * 3 + 1];
    float dz = sh_r[a * 3 + 2] - sh_r[b * 3 + 2];
    float d  = sqrtf(dx * dx + dy * dy + dz * dz + 1e-12f);
    float pm = sh_mask[a] * sh_mask[b];
    sh_d[idx] = (pm > 0.f) ? d : 1e18f;
  }
  __syncthreads();

  const int m0   = (wave & 3) * 16;     // M tile (atoms a)
  const int ng   = (wave >> 2) * 4;     // N tile group (features)
  const int lrow = lane & 15;

  for (int f = 0; f < NF; ++f) {
    const float mu = sh_mu[f], isig = sh_isig[f];
    #pragma unroll
    for (int q = 0; q < (MA * MA) / 512; ++q) {
      int idx = (q * 256 + tid) * 2;    // even -> 4B-aligned packed store
      float d0 = sh_d[idx]     - mu;
      float d1 = sh_d[idx + 1] - mu;
      *(unsigned*)(sh_w + idx) =
          pack2bf(5.f * __expf(-d0 * d0 * isig), 5.f * __expf(-d1 * d1 * isig));
    }
    __syncthreads();

    v8f acc[4] = {};
    #pragma unroll
    for (int k0 = 0; k0 < MA; k0 += 32) {
      v16bf afrag = load_frag(sh_w + (m0 + lrow) * MA + k0, lane);
      #pragma unroll
      for (int j = 0; j < 4; ++j) {
        int n0 = (ng + j) * 16;
        v16bf bfrag = load_frag(sh_ht + (n0 + lrow) * MA + k0, lane);
        acc[j] = __builtin_amdgcn_wmma_f32_16x16x32_bf16(
            false, afrag, false, bfrag, (short)0, acc[j], false, false);
      }
    }

    const int rhalf = (lane >> 4) << 3;
    #pragma unroll
    for (int j = 0; j < 4; ++j) {
      int col = FEAT + f * FEAT + (ng + j) * 16 + lrow;  // u part: 128 + f*128 + c
      #pragma unroll
      for (int i = 0; i < 8; ++i) {
        int row = m0 + i + rhalf;
        vbuf[((size_t)n * MA + row) * DIN + col] = f2bf(acc[j][i]);
      }
    }
    __syncthreads();   // protect sh_w before next filter rewrites it
  }
}

// ---- kernel 2: layer 1 GEMM (32768x2176 @ 2176x512) + SiLU -> bf16 t ------
// Block tile 256x128, 8 waves, each wave owns TWO 16-row M tiles so every
// B fragment feeds two WMMAs. Tiles staged via async global->LDS (no VGPRs).

__global__ void mlp1(const unsigned short* __restrict__ vbuf,
                     const unsigned short* __restrict__ w1t,   // transposed (H1 x DIN)
                     const float* __restrict__ b1,
                     unsigned short* __restrict__ tbuf) {
  __shared__ __align__(16) unsigned short shA [256 * 32];   // [row][k]
  __shared__ __align__(16) unsigned short shBt[128 * 32];   // [col][k] (transposed)
  const int rb = blockIdx.x, cb = blockIdx.y;
  const int tid = threadIdx.x, wave = tid >> 5, lane = tid & 31;
  const int row0 = rb * 256, col0 = cb * 128;
  const int m0 = wave * 32, lrow = lane & 15;

  const unsigned shA0  = lds_off(shA);
  const unsigned shBt0 = lds_off(shBt);

  v8f acc0[8] = {}, acc1[8] = {};
  for (int k0 = 0; k0 < DIN; k0 += 32) {
    __syncthreads();
    #pragma unroll
    for (int q = 0; q < 4; ++q) {                       // A: 256x32 bf16 async fill
      int li = q * 256 + tid;
      int rr = li >> 2, seg = (li & 3) * 16;
      async_b128(shA0 + (unsigned)(rr * 64 + seg),
                 (unsigned)(((size_t)(row0 + rr) * DIN + k0) * 2) + seg, vbuf);
    }
    #pragma unroll
    for (int q = 0; q < 2; ++q) {                       // Bt: 128x32 bf16 async fill
      int li = q * 256 + tid;
      int nn = li >> 2, seg = (li & 3) * 16;
      async_b128(shBt0 + (unsigned)(nn * 64 + seg),
                 (unsigned)(((size_t)(col0 + nn) * DIN + k0) * 2) + seg, w1t);
    }
    if (k0 + 32 < DIN) {                                // prefetch next K tile
      __builtin_prefetch(vbuf + (size_t)(row0 + tid) * DIN + k0 + 32, 0, 0);
      __builtin_prefetch(w1t + (size_t)(col0 + (tid >> 1)) * DIN + k0 + 32, 0, 0);
    }
    wait_async0();
    __syncthreads();

    v16bf af0 = load_frag(shA + (m0 + lrow) * 32, lane);
    v16bf af1 = load_frag(shA + (m0 + 16 + lrow) * 32, lane);
    #pragma unroll
    for (int j = 0; j < 8; ++j) {
      v16bf bfrag = load_frag(shBt + (j * 16 + lrow) * 32, lane);
      acc0[j] = __builtin_amdgcn_wmma_f32_16x16x32_bf16(
          false, af0, false, bfrag, (short)0, acc0[j], false, false);
      acc1[j] = __builtin_amdgcn_wmma_f32_16x16x32_bf16(
          false, af1, false, bfrag, (short)0, acc1[j], false, false);
    }
  }

  const int rhalf = (lane >> 4) << 3;
  #pragma unroll
  for (int j = 0; j < 8; ++j) {
    int col = col0 + j * 16 + lrow;
    float bias = b1[col];
    #pragma unroll
    for (int i = 0; i < 8; ++i) {
      int row = row0 + m0 + i + rhalf;
      float x = acc0[j][i] + bias;
      tbuf[(size_t)row * H1 + col] = f2bf(x / (1.f + __expf(-x)));   // SiLU
      int row2 = row + 16;
      float y = acc1[j][i] + bias;
      tbuf[(size_t)row2 * H1 + col] = f2bf(y / (1.f + __expf(-y)));
    }
  }
}

// ---- kernel 3: layer 2 GEMM (32768x512 @ 512x128) + residual ---------------

__global__ void mlp2(const unsigned short* __restrict__ tbuf,
                     const unsigned short* __restrict__ w2t,   // transposed (FEAT x H1)
                     const float* __restrict__ b2,
                     const float* __restrict__ h,
                     const int*   __restrict__ z,
                     float* __restrict__ out_h) {
  __shared__ __align__(16) unsigned short shA [256 * 32];
  __shared__ __align__(16) unsigned short shBt[128 * 32];
  const int rb = blockIdx.x;
  const int tid = threadIdx.x, wave = tid >> 5, lane = tid & 31;
  const int row0 = rb * 256;
  const int m0 = wave * 32, lrow = lane & 15;

  const unsigned shA0  = lds_off(shA);
  const unsigned shBt0 = lds_off(shBt);

  v8f acc0[8] = {}, acc1[8] = {};
  for (int k0 = 0; k0 < H1; k0 += 32) {
    __syncthreads();
    #pragma unroll
    for (int q = 0; q < 4; ++q) {
      int li = q * 256 + tid;
      int rr = li >> 2, seg = (li & 3) * 16;
      async_b128(shA0 + (unsigned)(rr * 64 + seg),
                 (unsigned)(((size_t)(row0 + rr) * H1 + k0) * 2) + seg, tbuf);
    }
    #pragma unroll
    for (int q = 0; q < 2; ++q) {
      int li = q * 256 + tid;
      int nn = li >> 2, seg = (li & 3) * 16;
      async_b128(shBt0 + (unsigned)(nn * 64 + seg),
                 (unsigned)(((size_t)nn * H1 + k0) * 2) + seg, w2t);
    }
    if (k0 + 32 < H1) {
      __builtin_prefetch(tbuf + (size_t)(row0 + tid) * H1 + k0 + 32, 0, 0);
    }
    wait_async0();
    __syncthreads();

    v16bf af0 = load_frag(shA + (m0 + lrow) * 32, lane);
    v16bf af1 = load_frag(shA + (m0 + 16 + lrow) * 32, lane);
    #pragma unroll
    for (int j = 0; j < 8; ++j) {
      v16bf bfrag = load_frag(shBt + (j * 16 + lrow) * 32, lane);
      acc0[j] = __builtin_amdgcn_wmma_f32_16x16x32_bf16(
          false, af0, false, bfrag, (short)0, acc0[j], false, false);
      acc1[j] = __builtin_amdgcn_wmma_f32_16x16x32_bf16(
          false, af1, false, bfrag, (short)0, acc1[j], false, false);
    }
  }

  const int rhalf = (lane >> 4) << 3;
  #pragma unroll
  for (int j = 0; j < 8; ++j) {
    int col = j * 16 + lrow;
    float bias = b2[col];
    #pragma unroll
    for (int i = 0; i < 8; ++i) {
      int row = row0 + m0 + i + rhalf;
      float m = (z[row] > -1) ? UPD : 0.f;
      out_h[(size_t)row * FEAT + col] =
          h[(size_t)row * FEAT + col] + (acc0[j][i] + bias) * m;
      int row2 = row + 16;
      float m2 = (z[row2] > -1) ? UPD : 0.f;
      out_h[(size_t)row2 * FEAT + col] =
          h[(size_t)row2 * FEAT + col] + (acc1[j][i] + bias) * m2;
    }
  }
}

// ---- launch ----------------------------------------------------------------

extern "C" void kernel_launch(void* const* d_in, const int* in_sizes, int n_in,
                              void* d_out, int out_size, void* d_ws, size_t ws_size,
                              hipStream_t stream) {
  const int*   z     = (const int*)d_in[0];
  const float* r     = (const float*)d_in[1];
  const float* h     = (const float*)d_in[2];
  const float* dist  = (const float*)d_in[3];
  const float* width = (const float*)d_in[4];
  const float* W1    = (const float*)d_in[5];
  const float* b1    = (const float*)d_in[6];
  const float* W2    = (const float*)d_in[7];
  const float* b2    = (const float*)d_in[8];

  // outputs concatenated in return order: z (int bits), r, h_new
  float* out   = (float*)d_out;
  float* out_r = out + (size_t)NMOL * MA;
  float* out_h = out_r + (size_t)NMOL * MA * 3;

  // workspace: v bf16 (142.6MB) | t bf16 (32MB) | W1t bf16 | W2t bf16
  unsigned short* vbuf = (unsigned short*)d_ws;
  unsigned short* tbuf = vbuf + (size_t)NMOL * MA * DIN;
  unsigned short* w1t  = tbuf + (size_t)NMOL * MA * H1;
  unsigned short* w2t  = w1t + (size_t)DIN * H1;

  hipMemcpyAsync(d_out, d_in[0], sizeof(int) * NMOL * MA,
                 hipMemcpyDeviceToDevice, stream);
  hipMemcpyAsync(out_r, d_in[1], sizeof(float) * NMOL * MA * 3,
                 hipMemcpyDeviceToDevice, stream);

  long nconv = (long)DIN * H1 + (long)H1 * FEAT;
  convert_weights<<<(int)((nconv + 255) / 256), 256, 0, stream>>>(W1, W2, w1t, w2t);
  rbf_agg<<<NMOL, 256, 0, stream>>>(z, r, h, dist, width, vbuf);
  mlp1<<<dim3((NMOL * MA) / 256, H1 / 128), 256, 0, stream>>>(vbuf, w1t, b1, tbuf);
  mlp2<<<(NMOL * MA) / 256, 256, 0, stream>>>(tbuf, w2t, b2, h, z, out_h);
}
